// BitConv2d_36885179138368
// MI455X (gfx1250) — 8-bit matrix operand order repaired and checked by reading, compiled, not run on hardware
//
#include <hip/hip_runtime.h>

typedef __attribute__((ext_vector_type(8))) int v8i;
typedef __attribute__((ext_vector_type(4))) int v4i;
typedef __attribute__((ext_vector_type(4))) unsigned int u4i;

#define EPSQ 1e-5f
constexpr int Nb = 16, Cc = 64, Hh = 256, Ww = 256;
constexpr int HW = Hh * Ww;

constexpr size_t WS_WQ = 256;
constexpr size_t WS_QX = 37120;
constexpr size_t WS_RS = 67145984;

#if defined(__has_builtin)
#if __has_builtin(__builtin_amdgcn_tensor_load_to_lds)
#define HAVE_TDM 1
#endif
#endif
#ifndef HAVE_TDM
#define HAVE_TDM 0
#endif

__global__ __launch_bounds__(256) void act_quant_kernel(
    const float* __restrict__ x, signed char* __restrict__ qx,
    float* __restrict__ rs) {
  int pid = blockIdx.x * 256 + threadIdx.x;
  int n = pid >> 16;
  int hw = pid & 65535;
  const float* xp = x + (size_t)n * Cc * HW + hw;

  float xv[64];
  float amax = 0.0f;
#pragma unroll
  for (int c = 0; c < 64; ++c) {
    float v = xp[(size_t)c * HW];
    xv[c] = v;
    amax = fmaxf(amax, fabsf(v));
  }
  float m = fmaxf(amax, EPSQ);
  float scale = 127.0f / m;
  rs[pid] = m / 127.0f;

  int dws[16];
#pragma unroll
  for (int j = 0; j < 16; ++j) {
    int d = 0;
#pragma unroll
    for (int b = 0; b < 4; ++b) {
      float q = rintf(xv[4 * j + b] * scale);
      q = fminf(fmaxf(q, -128.0f), 127.0f);
      d |= (((int)q) & 0xFF) << (8 * b);
    }
    dws[j] = d;
  }
  v4i* op = (v4i*)(qx + (size_t)pid * 64);
#pragma unroll
  for (int j = 0; j < 4; ++j) {
    v4i v = {dws[4 * j + 0], dws[4 * j + 1], dws[4 * j + 2], dws[4 * j + 3]};
    op[j] = v;
  }
}

__global__ __launch_bounds__(256) void wquant_kernel(
    const float* __restrict__ w, float* __restrict__ hdr,
    int* __restrict__ wq) {
  __shared__ float red[256];
  int tid = threadIdx.x;
  float s = 0.0f;
  for (int i = tid; i < 36864; i += 256) s += fabsf(w[i]);
  red[tid] = s;
  __syncthreads();
  for (int off = 128; off > 0; off >>= 1) {
    if (tid < off) red[tid] += red[tid + off];
    __syncthreads();
  }
  float m = fmaxf(red[0] / 36864.0f, EPSQ);
  if (tid == 0) hdr[0] = m;
  float scale = 1.0f / m;

  for (int idx = tid; idx < 9216; idx += 256) {
    int v = idx & 7;
    int lane = (idx >> 3) & 31;
    int cg = (idx >> 8) & 3;
    int tap = idx >> 10;
    int hl = lane >> 4;
    int cout = cg * 16 + (lane & 15);
    int kh = tap / 3, kw = tap % 3;
    int d = 0;
#pragma unroll
    for (int b = 0; b < 4; ++b) {
      int K = ((v >> 2) * 32) + hl * 16 +
              ((v & 3) * 4) + b;
      float wv = w[((cout * 64 + K) * 3 + kh) * 3 + kw];
      float q = rintf(wv * scale);
      q = fminf(fmaxf(q, -1.0f), 1.0f);
      d |= (((int)q) & 0xFF) << (8 * b);
    }
    wq[idx] = d;
  }
}

__global__ __launch_bounds__(256) void bitconv_wmma_kernel(
    const signed char* __restrict__ qx, const float* __restrict__ rs,
    const v4i* __restrict__ wq4, const float* __restrict__ hdr,
    const float* __restrict__ bias, float* __restrict__ out) {
  __shared__ v4i ldsW[2304];
  int tid = threadIdx.x;

#if HAVE_TDM
  if (tid < 32) {
    unsigned long long gaddr = (unsigned long long)(const void*)wq4;
    unsigned int laddr = (unsigned int)(unsigned long long)(void*)ldsW;
    u4i g0;
    g0.x = 1u;
    g0.y = laddr;
    g0.z = (unsigned int)(gaddr & 0xFFFFFFFFu);
    g0.w = (unsigned int)(gaddr >> 32) | (2u << 30);
    v8i g1;
    g1[0] = 3 << 16;
    g1[1] = 0x1200 << 16;
    g1[2] = 1 << 16;
    g1[3] = 0x1200 << 16;
    g1[4] = 1;
    g1[5] = 0x1200;
    g1[6] = 0x1200 << 16;
    g1[7] = 0;
    v4i gz = {};
#if __clang_major__ >= 23
    v8i gz8 = {};
    __builtin_amdgcn_tensor_load_to_lds(g0, g1, gz, gz, gz8, 0);
#else
    __builtin_amdgcn_tensor_load_to_lds(g0, g1, gz, gz, 0);
#endif
    __builtin_amdgcn_s_wait_tensorcnt(0);
  }
  __syncthreads();
#else
#pragma unroll
  for (int i = 0; i < 9; ++i) ldsW[tid + i * 256] = wq4[tid + i * 256];
  __syncthreads();
#endif

  int lane = tid & 31;
  int wave = tid >> 5;
  int hl = lane >> 4;
  int p = lane & 15;
  int n = blockIdx.z;
  int h = blockIdx.y * 8 + wave;
  int w0 = blockIdx.x * 16;

  float accf[4][8];
#pragma unroll
  for (int cg = 0; cg < 4; ++cg)
#pragma unroll
    for (int v = 0; v < 8; ++v) accf[cg][v] = 0.0f;

#pragma unroll
  for (int tap = 0; tap < 9; ++tap) {
    int dh = tap / 3 - 1;
    int dw = tap % 3 - 1;
    int hy = h + dh;
    int wx = w0 + p + dw;
    bool valid = (hy >= 0) && (hy < Hh) && (wx >= 0) && (wx < Ww);
    int hyc = min(max(hy, 0), Hh - 1);
    int wxc = min(max(wx, 0), Ww - 1);
    int pix = (n * Hh + hyc) * Ww + wxc;

    const v4i* bp = (const v4i*)(qx + (size_t)pix * 64 + (hl << 4));
    v4i z = {};
    v4i b0 = bp[0];
    v4i b1 = bp[2];
    b0 = valid ? b0 : z;
    b1 = valid ? b1 : z;
    float sc = valid ? rs[pix] : 0.0f;
    v8i B = __builtin_shufflevector(b0, b1, 0, 1, 2, 3, 4, 5, 6, 7);

#pragma unroll
    for (int cg = 0; cg < 4; ++cg) {
      const v4i* ap = &ldsW[(tap * 4 + cg) * 64 + lane * 2];
      v4i a0 = ap[0];
      v4i a1 = ap[1];
      v8i A = __builtin_shufflevector(a0, a1, 0, 1, 2, 3, 4, 5, 6, 7);
      v8i c0 = {};
      v8i d = __builtin_amdgcn_wmma_i32_16x16x64_iu8(true, A, true, B, c0,
                                                     false, false);
#pragma unroll
      for (int v = 0; v < 8; ++v) accf[cg][v] += (float)d[v] * sc;
    }
  }

  float m = hdr[0];
#pragma unroll
  for (int cg = 0; cg < 4; ++cg) {
#pragma unroll
    for (int v = 0; v < 8; ++v) {
      int cout = cg * 16 + hl * 8 + v;
      float o = accf[cg][v] * m + bias[cout];
      out[(((size_t)n * Cc + cout) * Hh + h) * Ww + w0 + p] = o;
    }
  }
}

extern "C" void kernel_launch(void* const* d_in, const int* in_sizes, int n_in,
                              void* d_out, int out_size, void* d_ws,
                              size_t ws_size, hipStream_t stream) {
  const float* x = (const float*)d_in[0];
  const float* w = (const float*)d_in[1];
  const float* bias = (const float*)d_in[2];
  float* out = (float*)d_out;

  char* ws = (char*)d_ws;
  float* hdr = (float*)ws;
  int* wq = (int*)(ws + WS_WQ);
  signed char* qx = (signed char*)(ws + WS_QX);
  float* rs = (float*)(ws + WS_RS);

  act_quant_kernel<<<4096, 256, 0, stream>>>(x, qx, rs);
  wquant_kernel<<<1, 256, 0, stream>>>(w, hdr, wq);
  bitconv_wmma_kernel<<<dim3(Ww / 16, Hh / 8, Nb), 256, 0, stream>>>(
      qx, rs, (const v4i*)wq, hdr, bias, out);
}
